// PolicyNet_27376121545029
// MI455X (gfx1250) — matrix operand order repaired and checked by reading, not compiled, not run on hardware
//
#include <hip/hip_runtime.h>

// Problem constants (match reference)
#define NN   50000
#define EE   800000
#define EPN  850000   // edges + self loops
#define GGR  50
#define NPG  1000     // nodes per graph (contiguous batch)

typedef __attribute__((ext_vector_type(16))) __bf16         v16bf;
typedef __attribute__((ext_vector_type(8)))  unsigned int   v8u;
typedef __attribute__((ext_vector_type(4)))  unsigned int   v4u;
typedef __attribute__((ext_vector_type(8)))  float          v8f;

__device__ __forceinline__ unsigned short bfbits(float x) {
    return __builtin_bit_cast(unsigned short, (__bf16)x);   // hw cvt, RNE
}
// order-preserving float<->uint encoding for atomicMax-based segment_max
__device__ __forceinline__ unsigned fenc(float f) {
    unsigned u = __float_as_uint(f);
    return (u & 0x80000000u) ? ~u : (u | 0x80000000u);
}
__device__ __forceinline__ float fdec(unsigned u) {
    unsigned b = (u & 0x80000000u) ? (u & 0x7FFFFFFFu) : ~u;
    return __uint_as_float(b);
}

// ---------------- weight convert + transpose: f32[K,N] -> bf16[N,K] ----------
__global__ void k_cvt_t(const float* __restrict__ in, unsigned short* __restrict__ out,
                        int K, int Nc) {
    int tid = blockIdx.x * blockDim.x + threadIdx.x;
    if (tid >= K * Nc) return;
    int k = tid / Nc, n = tid % Nc;
    out[(size_t)n * K + k] = bfbits(in[tid]);
}

// ---------------- row-major f32 -> bf16 (same layout), 4 elems/thread --------
__global__ void k_cvt_row(const float* __restrict__ in, unsigned short* __restrict__ out,
                          int total) {
    int i = (blockIdx.x * blockDim.x + threadIdx.x) * 4;
    if (i >= total) return;
    float4 v = *(const float4*)(in + i);
    out[i + 0] = bfbits(v.x);
    out[i + 1] = bfbits(v.y);
    out[i + 2] = bfbits(v.z);
    out[i + 3] = bfbits(v.w);
}

// ---------------- bf16 WMMA GEMM: C[M,N] = A[M,K] @ B[K,N] (+bias, relu) -----
// A: bf16 [M,K] row-major. Bt: bf16 [N,K]. Each wave: one 16-row strip x two
// 16-col tiles (A fragment reused across 2 WMMAs). K compile-time -> unrolled.
template <int K>
__global__ __launch_bounds__(128) void k_gemm(const unsigned short* __restrict__ A,
                                              const unsigned short* __restrict__ Bt,
                                              const float* __restrict__ bias,
                                              float* __restrict__ C,
                                              unsigned short* __restrict__ Cbf,
                                              int M, int Nc, int relu) {
    int lane  = threadIdx.x & 31;
    int wave  = threadIdx.x >> 5;
    int mtile = blockIdx.x * 4 + wave;
    if (mtile * 16 >= M) return;            // wave-uniform: EXEC stays all-ones
    int mbase = mtile * 16;
    int nbase = blockIdx.y * 32;            // two 16-wide N tiles per wave

    int row_a = mbase + (lane & 15);
    int col0  = nbase + (lane & 15);
    int col1  = col0 + 16;
    int koff  = (lane >> 4) << 4;           // A half-wave K group: 0 or 16
    int bkoff = (lane >> 4) << 4;           // B half-wave K group: 0 or 16

    const unsigned short* arow  = A  + (size_t)row_a * K;
    const unsigned short* brow0 = Bt + (size_t)col0 * K;
    const unsigned short* brow1 = Bt + (size_t)col1 * K;

    v8f acc0 = {}, acc1 = {};
#pragma unroll
    for (int k0 = 0; k0 < K; k0 += 32) {
        v4u alo = *(const v4u*)(arow + k0 + koff);
        v4u ahi = *(const v4u*)(arow + k0 + 8 + koff);
        v8u a8;
#pragma unroll
        for (int i = 0; i < 4; i++) { a8[i] = alo[i]; a8[4 + i] = ahi[i]; }
        v16bf afrag = __builtin_bit_cast(v16bf, a8);
        v16bf b0 = __builtin_bit_cast(v16bf, *(const v8u*)(brow0 + k0 + bkoff));
        v16bf b1 = __builtin_bit_cast(v16bf, *(const v8u*)(brow1 + k0 + bkoff));
        acc0 = __builtin_amdgcn_wmma_f32_16x16x32_bf16(false, afrag, false, b0,
                                                       (short)0, acc0, false, false);
        acc1 = __builtin_amdgcn_wmma_f32_16x16x32_bf16(false, afrag, false, b1,
                                                       (short)0, acc1, false, false);
    }
    float bv0 = bias ? bias[col0] : 0.f;
    float bv1 = bias ? bias[col1] : 0.f;
    int   mrow = mbase + ((lane >> 4) << 3);
#pragma unroll
    for (int r = 0; r < 8; r++) {
        float v0 = acc0[r] + bv0;
        float v1 = acc1[r] + bv1;
        if (relu) { v0 = fmaxf(v0, 0.f); v1 = fmaxf(v1, 0.f); }
        size_t o = (size_t)(mrow + r) * Nc;
        if (C)   { C[o + col0] = v0; C[o + col1] = v1; }
        if (Cbf) { Cbf[o + col0] = bfbits(v0); Cbf[o + col1] = bfbits(v1); }
    }
}

// ---------------- small init kernels -----------------------------------------
__global__ void k_fillf(float* p, float v, int n) {
    int i = blockIdx.x * blockDim.x + threadIdx.x;
    if (i < n) p[i] = v;
}
__global__ void k_fillu(unsigned* p, unsigned v, int n) {
    int i = blockIdx.x * blockDim.x + threadIdx.x;
    if (i < n) p[i] = v;
}
__global__ void k_binit(float* agg, const float* __restrict__ bias, int total, int F) {
    int i = blockIdx.x * blockDim.x + threadIdx.x;
    if (i < total) agg[i] = bias[i % F];
}

// ---------------- attention logits per (node, head) --------------------------
__global__ void k_attn_logits(const float* __restrict__ xh,
                              const float* __restrict__ asrc,
                              const float* __restrict__ adst,
                              float* __restrict__ alsrc, float* __restrict__ aldst) {
    int tid = blockIdx.x * blockDim.x + threadIdx.x;
    if (tid >= NN * 4) return;
    int n = tid >> 2, h = tid & 3;
    const float* xr = xh + (size_t)n * 256 + h * 64;
    const float* as = asrc + h * 64;
    const float* ad = adst + h * 64;
    float s = 0.f, d = 0.f;
#pragma unroll 16
    for (int i = 0; i < 64; i++) { float v = xr[i]; s += v * as[i]; d += v * ad[i]; }
    alsrc[tid] = s;
    aldst[tid] = d;
}

// ---------------- edge pass A: leaky-relu logits + segment max ---------------
__global__ void k_edge_A(const int* __restrict__ src, const int* __restrict__ dst,
                         const float* __restrict__ alsrc, const float* __restrict__ aldst,
                         float* __restrict__ al, unsigned* __restrict__ amaxu) {
    int e = blockIdx.x * blockDim.x + threadIdx.x;
    if (e >= EPN) return;
    int s, d;
    if (e < EE) { s = src[e]; d = dst[e]; } else { s = d = e - EE; }
#pragma unroll
    for (int h = 0; h < 4; h++) {
        float a = alsrc[s * 4 + h] + aldst[d * 4 + h];
        a = (a > 0.f) ? a : 0.2f * a;
        al[(size_t)e * 4 + h] = a;
        atomicMax(&amaxu[d * 4 + h], fenc(a));
    }
}

// ---------------- edge pass B1: exp(al - max), segment sum (den) -------------
__global__ void k_edge_B1(const int* __restrict__ src, const int* __restrict__ dst,
                          float* __restrict__ al, const unsigned* __restrict__ amaxu,
                          float* __restrict__ den) {
    int e = blockIdx.x * blockDim.x + threadIdx.x;
    if (e >= EPN) return;
    int d = (e < EE) ? dst[e] : (e - EE);
#pragma unroll
    for (int h = 0; h < 4; h++) {
        float ex = expf(al[(size_t)e * 4 + h] - fdec(amaxu[d * 4 + h]));
        al[(size_t)e * 4 + h] = ex;
        atomicAdd(&den[d * 4 + h], ex);
    }
}

// ---------------- edge pass B2: weighted scatter-add of xh[src] --------------
__global__ void k_edge_B2(const int* __restrict__ src, const int* __restrict__ dst,
                          const float* __restrict__ ex, const float* __restrict__ den,
                          const float* __restrict__ xh, float* __restrict__ agg) {
    int tid = blockIdx.x * blockDim.x + threadIdx.x;
    if (tid >= EPN * 4) return;
    int e = tid >> 2, h = tid & 3;
    int s, d;
    if (e < EE) { s = src[e]; d = dst[e]; } else { s = d = e - EE; }
    float w = ex[(size_t)e * 4 + h] / (den[d * 4 + h] + 1e-16f);
    const float* xr = xh + (size_t)s * 256 + h * 64;
    float*       ar = agg + (size_t)d * 256 + h * 64;
#pragma unroll
    for (int i = 0; i < 64; i += 4) {
        float4 v = *(const float4*)(xr + i);
        atomicAdd(ar + i + 0, v.x * w);
        atomicAdd(ar + i + 1, v.y * w);
        atomicAdd(ar + i + 2, v.z * w);
        atomicAdd(ar + i + 3, v.w * w);
    }
}

// ---------------- graph LayerNorm (mode='graph'); optional residual/relu/bf16
__global__ void k_graph_ln(const float* __restrict__ xin, const float* __restrict__ res,
                           const float* __restrict__ w, const float* __restrict__ b,
                           float* __restrict__ out, unsigned short* __restrict__ outbf,
                           int F, int relu) {
    int g = blockIdx.x, t = threadIdx.x;
    size_t base  = (size_t)g * NPG * F;
    int    total = NPG * F;
    float s = 0.f, ss = 0.f;
    for (int i = t; i < total; i += 256) {
        float v = xin[base + i] + (res ? res[base + i] : 0.f);
        s += v; ss += v * v;
    }
    __shared__ float sh1[256], sh2[256];
    __shared__ float smean, sinv;
    sh1[t] = s; sh2[t] = ss;
    __syncthreads();
    for (int off = 128; off > 0; off >>= 1) {
        if (t < off) { sh1[t] += sh1[t + off]; sh2[t] += sh2[t + off]; }
        __syncthreads();
    }
    if (t == 0) {
        float m   = sh1[0] / (float)total;
        float var = sh2[0] / (float)total - m * m;
        smean = m;
        sinv  = rsqrtf(var + 1e-5f);
    }
    __syncthreads();
    float m = smean, inv = sinv;
    for (int i = t; i < total; i += 256) {
        int   f = i % F;
        float v = xin[base + i] + (res ? res[base + i] : 0.f);
        v = (v - m) * inv * w[f] + b[f];
        if (relu) v = fmaxf(v, 0.f);
        out[base + i] = v;
        if (outbf) outbf[base + i] = bfbits(v);
    }
}

// ---------------- per-graph mean pool (F=64) ---------------------------------
__global__ void k_mean_pool(const float* __restrict__ x, float* __restrict__ mean) {
    int g = blockIdx.x, t = threadIdx.x;
    int f = t & 63, stripe = t >> 6;   // 4 stripes
    float s = 0.f;
    for (int n = stripe; n < NPG; n += 4) s += x[((size_t)g * NPG + n) * 64 + f];
    __shared__ float sh[256];
    sh[t] = s;
    __syncthreads();
    if (t < 64) mean[g * 64 + t] = (sh[t] + sh[t + 64] + sh[t + 128] + sh[t + 192]) * (1.f / NPG);
}

// ---------------- concat [x | mean(graph) | x1] -> bf16 [N,192] --------------
__global__ void k_concat(const float* __restrict__ x, const float* __restrict__ mean,
                         const float* __restrict__ x1, unsigned short* __restrict__ xc) {
    int tid = blockIdx.x * blockDim.x + threadIdx.x;
    if (tid >= NN * 192) return;
    int n = tid / 192, j = tid % 192;
    float v;
    if (j < 64)       v = x[(size_t)n * 64 + j];
    else if (j < 128) v = mean[(n / NPG) * 64 + (j - 64)];
    else              v = x1[(size_t)n * 64 + (j - 128)];
    xc[tid] = bfbits(v);
}

// ---------------- final 256 -> 2 projection ----------------------------------
__global__ void k_out_proj(const float* __restrict__ p, const float* __restrict__ W,
                           float* __restrict__ p2) {
    int n = blockIdx.x * blockDim.x + threadIdx.x;
    if (n >= NN) return;
    const float* pr = p + (size_t)n * 256;
    float s0 = 0.f, s1 = 0.f;
#pragma unroll 8
    for (int k = 0; k < 256; k++) { float v = pr[k]; s0 += v * W[2 * k]; s1 += v * W[2 * k + 1]; }
    p2[n * 2 + 0] = s0;
    p2[n * 2 + 1] = s1;
}

// ---------------- per-graph softmax over 2000 logits -------------------------
__global__ void k_softmax(const float* __restrict__ p2, float* __restrict__ out) {
    int g = blockIdx.x, t = threadIdx.x;
    const int rowlen = 2 * NPG;
    const float* row = p2 + (size_t)g * rowlen;
    __shared__ float sh[256];
    __shared__ float smax, ssum;
    float mx = -3.402823466e+38f;
    for (int i = t; i < rowlen; i += 256) mx = fmaxf(mx, row[i]);
    sh[t] = mx; __syncthreads();
    for (int off = 128; off > 0; off >>= 1) { if (t < off) sh[t] = fmaxf(sh[t], sh[t + off]); __syncthreads(); }
    if (t == 0) smax = sh[0];
    __syncthreads();
    mx = smax;
    float s = 0.f;
    for (int i = t; i < rowlen; i += 256) s += expf(row[i] - mx);
    sh[t] = s; __syncthreads();
    for (int off = 128; off > 0; off >>= 1) { if (t < off) sh[t] += sh[t + off]; __syncthreads(); }
    if (t == 0) ssum = sh[0];
    __syncthreads();
    float inv = 1.f / ssum;
    for (int i = t; i < rowlen; i += 256) out[(size_t)g * rowlen + i] = expf(row[i] - mx) * inv;
}

// =============================================================================
extern "C" void kernel_launch(void* const* d_in, const int* in_sizes, int n_in,
                              void* d_out, int out_size, void* d_ws, size_t ws_size,
                              hipStream_t stream) {
    (void)in_sizes; (void)n_in; (void)out_size; (void)ws_size;

    const float* node_attr = (const float*)d_in[0];
    const float* yinp      = (const float*)d_in[1];
    const int*   ei        = (const int*)d_in[2];
    const int*   src       = ei;
    const int*   dst       = ei + EE;
    const float* enc_W  = (const float*)d_in[4];
    const float* enc_b  = (const float*)d_in[5];
    const float* gat_W  = (const float*)d_in[6];
    const float* att_s  = (const float*)d_in[7];
    const float* att_d  = (const float*)d_in[8];
    const float* gat_b  = (const float*)d_in[9];
    const float* proj_W = (const float*)d_in[10];
    const float* ln1w   = (const float*)d_in[11];
    const float* ln1b   = (const float*)d_in[12];
    const float* ffW1   = (const float*)d_in[13];
    const float* ffb1   = (const float*)d_in[14];
    const float* ffW2   = (const float*)d_in[15];
    const float* ffb2   = (const float*)d_in[16];
    const float* ln2w   = (const float*)d_in[17];
    const float* ln2b   = (const float*)d_in[18];
    const float* polW   = (const float*)d_in[19];
    const float* pol_as = (const float*)d_in[20];
    const float* pol_ad = (const float*)d_in[21];
    const float* pol_b  = (const float*)d_in[22];
    const float* pol_lnw  = (const float*)d_in[23];
    const float* pol_lnb  = (const float*)d_in[24];
    const float* pol_outW = (const float*)d_in[25];

    // ---- workspace carve-up (256B aligned) ----
    char*  ws  = (char*)d_ws;
    size_t off = 0;
    auto alloc = [&](size_t bytes) -> void* {
        void* p = ws + off;
        off = (off + bytes + 255) & ~(size_t)255;
        return p;
    };
    float*    f_x   = (float*)alloc((size_t)NN * 64 * 4);
    float*    f_x1  = (float*)alloc((size_t)NN * 64 * 4);
    float*    f_sm  = (float*)alloc((size_t)NN * 64 * 4);
    float*    big1  = (float*)alloc((size_t)NN * 256 * 4);
    float*    big2  = (float*)alloc((size_t)NN * 256 * 4);
    float*    f_al  = (float*)alloc((size_t)EPN * 4 * 4);
    float*    f_as  = (float*)alloc((size_t)NN * 4 * 4);
    float*    f_ad  = (float*)alloc((size_t)NN * 4 * 4);
    float*    f_den = (float*)alloc((size_t)NN * 4 * 4);
    unsigned* u_amx = (unsigned*)alloc((size_t)NN * 4 * 4);
    float*    f_mean = (float*)alloc((size_t)GGR * 64 * 4);
    float*    f_p2   = (float*)alloc((size_t)NN * 2 * 4);
    unsigned short* bfb   = (unsigned short*)alloc((size_t)NN * 256 * 2); // bf16 staging (agg/ff1/xc)
    unsigned short* x_bf  = (unsigned short*)alloc((size_t)NN * 64 * 2);  // bf16 copy of x
    unsigned short* a_bf  = (unsigned short*)alloc((size_t)NN * 64 * 2);  // bf16 encoder input
    unsigned short* wt_enc = (unsigned short*)alloc(64 * 64 * 2);
    unsigned short* wt_gat[3], *wt_proj[3], *wt_ff1[3], *wt_ff2[3];
    for (int l = 0; l < 3; l++) {
        wt_gat[l]  = (unsigned short*)alloc(256 * 64 * 2);
        wt_proj[l] = (unsigned short*)alloc(64 * 256 * 2);
        wt_ff1[l]  = (unsigned short*)alloc(256 * 64 * 2);
        wt_ff2[l]  = (unsigned short*)alloc(64 * 256 * 2);
    }
    unsigned short* wt_pol = (unsigned short*)alloc(256 * 192 * 2);

    auto cvt = [&](const float* in, unsigned short* out, int K, int Nc) {
        k_cvt_t<<<(K * Nc + 255) / 256, 256, 0, stream>>>(in, out, K, Nc);
    };
    auto cvt_row = [&](const float* in, unsigned short* out, int total) {
        k_cvt_row<<<(total / 4 + 255) / 256, 256, 0, stream>>>(in, out, total);
    };
    auto gemm = [&](int K, const unsigned short* A, const unsigned short* Bt,
                    const float* bias, float* C, unsigned short* Cbf, int Nc, int relu) {
        dim3 grid((NN / 16 + 3) / 4, Nc / 32);
        if (K == 64)
            k_gemm<64><<<grid, 128, 0, stream>>>(A, Bt, bias, C, Cbf, NN, Nc, relu);
        else if (K == 192)
            k_gemm<192><<<grid, 128, 0, stream>>>(A, Bt, bias, C, Cbf, NN, Nc, relu);
        else
            k_gemm<256><<<grid, 128, 0, stream>>>(A, Bt, bias, C, Cbf, NN, Nc, relu);
    };

    // ---- convert all weights to bf16 [N,K] ----
    cvt(enc_W, wt_enc, 64, 64);
    for (int l = 0; l < 3; l++) {
        cvt(gat_W  + (size_t)l * 64 * 256, wt_gat[l],  64, 256);
        cvt(proj_W + (size_t)l * 256 * 64, wt_proj[l], 256, 64);
        cvt(ffW1   + (size_t)l * 64 * 256, wt_ff1[l],  64, 256);
        cvt(ffW2   + (size_t)l * 256 * 64, wt_ff2[l],  256, 64);
    }
    cvt(polW, wt_pol, 192, 256);

    // ---- encoder: x (f32 + bf16 copy), x1 (f32) ----
    cvt_row(node_attr, a_bf, NN * 64);
    gemm(64, a_bf, wt_enc, enc_b, f_x, x_bf, 64, 0);
    cvt_row(yinp, a_bf, NN * 64);
    gemm(64, a_bf, wt_enc, enc_b, f_x1, nullptr, 64, 0);

    const int ebA  = (EPN + 255) / 256;
    const int ebB2 = (EPN * 4 + 255) / 256;
    const int nh   = (NN * 4 + 255) / 256;

    // one full GAT aggregation: xh (f32) -> agg (f32, bias pre-initialized)
    auto gat_edges = [&](const float* xh, float* agg, const float* asrc,
                         const float* adst, const float* bias) {
        k_attn_logits<<<nh, 256, 0, stream>>>(xh, asrc, adst, f_as, f_ad);
        k_fillu<<<nh, 256, 0, stream>>>(u_amx, 0u, NN * 4);
        k_fillf<<<nh, 256, 0, stream>>>(f_den, 0.f, NN * 4);
        k_edge_A<<<ebA, 256, 0, stream>>>(src, dst, f_as, f_ad, f_al, u_amx);
        k_edge_B1<<<ebA, 256, 0, stream>>>(src, dst, f_al, u_amx, f_den);
        k_binit<<<(NN * 256 + 255) / 256, 256, 0, stream>>>(agg, bias, NN * 256, 256);
        k_edge_B2<<<ebB2, 256, 0, stream>>>(src, dst, f_al, f_den, xh, agg);
    };

    // ---- 3 transformer blocks ----
    for (int l = 0; l < 3; l++) {
        gemm(64, x_bf, wt_gat[l], nullptr, big1, nullptr, 256, 0);       // xh = x @ gat_W
        gat_edges(big1, big2, att_s + l * 256, att_d + l * 256, gat_b + l * 256);
        cvt_row(big2, bfb, NN * 256);                                    // agg -> bf16
        gemm(256, bfb, wt_proj[l], nullptr, f_sm, nullptr, 64, 0);       // h = agg @ proj
        k_graph_ln<<<GGR, 256, 0, stream>>>(f_x, f_sm, ln1w + l * 64, ln1b + l * 64,
                                            f_x, x_bf, 64, 0);           // x = LN(x+h)
        gemm(64, x_bf, wt_ff1[l], ffb1 + l * 256, nullptr, bfb, 256, 1); // relu(x@W1+b1) bf16
        gemm(256, bfb, wt_ff2[l], ffb2 + l * 64, f_sm, nullptr, 64, 0);  // @W2+b2
        k_graph_ln<<<GGR, 256, 0, stream>>>(f_x, f_sm, ln2w + l * 64, ln2b + l * 64,
                                            f_x, x_bf, 64, 0);           // x = LN(x+h)
    }

    // ---- policy head ----
    k_mean_pool<<<GGR, 256, 0, stream>>>(f_x, f_mean);
    k_concat<<<(NN * 192 + 255) / 256, 256, 0, stream>>>(f_x, f_mean, f_x1, bfb);
    gemm(192, bfb, wt_pol, nullptr, big2, nullptr, 256, 0);              // xh = x_c @ pol_W
    gat_edges(big2, big1, pol_as, pol_ad, pol_b);                        // agg -> big1
    k_graph_ln<<<GGR, 256, 0, stream>>>(big1, nullptr, pol_lnw, pol_lnb,
                                        big1, nullptr, 256, 1);          // relu(LN(p))
    k_out_proj<<<(NN + 255) / 256, 256, 0, stream>>>(big1, pol_outW, f_p2);
    k_softmax<<<GGR, 256, 0, stream>>>(f_p2, (float*)d_out);
}
